// Torch_17377437680120
// MI455X (gfx1250) — compile-verified, matrix operand order checked by reading, not run on hardware
//
#include <hip/hip_runtime.h>
#include <stdint.h>

// ---- CDNA5 WMMA vector types (wave32) ----
typedef __attribute__((ext_vector_type(16))) __bf16 v16bf;
typedef __attribute__((ext_vector_type(8)))  float  v8f;

// Problem constants (B=4, S=8192, IN=OUT=1024)
#define M_TOT 32768
#define N_DIM 1024
#define K_DIM 1024

// Tiling
#define BM 128
#define BN 64
#define BK 32
// LDS row stride in uints (16 data uints = 32 bf16, +4 pad -> 80B rows,
// 16B-aligned and bank-conflict-free for 16-lane b128 fragment loads)
#define ROW_U 20

__device__ __forceinline__ unsigned pack2_bf16(float a, float b) {
  // round-to-nearest-even fp32 -> bf16, packed pair
  unsigned ua = __float_as_uint(a);
  unsigned ub = __float_as_uint(b);
  ua += 0x7FFFu + ((ua >> 16) & 1u);
  ub += 0x7FFFu + ((ub >> 16) & 1u);
  return (ua >> 16) | (ub & 0xFFFF0000u);
}

union FragU {
  v16bf v;
  uint4 q[2];
};

__global__ __launch_bounds__(256)
void dual_gemm_masked_kernel(const float* __restrict__ x,
                             const uint8_t* __restrict__ visual_mask,
                             const float* __restrict__ Wv,
                             const float* __restrict__ Wt,
                             float* __restrict__ out) {
  __shared__ unsigned xs [BM * ROW_U];  // x tile, bf16 pairs
  __shared__ unsigned wvs[BN * ROW_U];  // W_v tile
  __shared__ unsigned wts[BN * ROW_U];  // W_t tile

  const int tid  = threadIdx.x;
  const int lane = tid & 31;
  const int wave = tid >> 5;

  const int m0 = blockIdx.y * BM;   // block row in flattened (B*S)
  const int n0 = blockIdx.x * BN;   // block col in OUT_DIM

  // 8 waves: 4 along M x 2 along N; each wave owns 32x32 of the output
  const int wm = (wave & 3) * 32;
  const int wn = (wave >> 2) * 32;

  // Staging coordinates: each thread loads float4 slices
  const int lrow = tid >> 3;        // 0..31
  const int lcol = (tid & 7) * 4;   // 0,4,...,28  (column within K-block)

  v8f accV[2][2] = {};
  v8f accT[2][2] = {};

  for (int k0 = 0; k0 < K_DIM; k0 += BK) {
    // Prefetch next K-slab of x into L2 (global_prefetch_b8)
    if (k0 + BK < K_DIM) {
      __builtin_prefetch(x + (size_t)(m0 + lrow) * K_DIM + (k0 + BK) + lcol, 0, 1);
    }

    // ---- stage x tile: 128x32 fp32 -> bf16 in LDS ----
#pragma unroll
    for (int it = 0; it < 4; ++it) {
      const int r = lrow + it * 32;
      const float4 v = *(const float4*)(x + (size_t)(m0 + r) * K_DIM + k0 + lcol);
      xs[r * ROW_U + (lcol >> 1)]     = pack2_bf16(v.x, v.y);
      xs[r * ROW_U + (lcol >> 1) + 1] = pack2_bf16(v.z, v.w);
    }
    // ---- stage both weight tiles: 64x32 fp32 -> bf16 in LDS ----
#pragma unroll
    for (int it = 0; it < 2; ++it) {
      const int r = lrow + it * 32;
      const float4 a = *(const float4*)(Wv + (size_t)(n0 + r) * K_DIM + k0 + lcol);
      wvs[r * ROW_U + (lcol >> 1)]     = pack2_bf16(a.x, a.y);
      wvs[r * ROW_U + (lcol >> 1) + 1] = pack2_bf16(a.z, a.w);
      const float4 b = *(const float4*)(Wt + (size_t)(n0 + r) * K_DIM + k0 + lcol);
      wts[r * ROW_U + (lcol >> 1)]     = pack2_bf16(b.x, b.y);
      wts[r * ROW_U + (lcol >> 1) + 1] = pack2_bf16(b.z, b.w);
    }
    __syncthreads();

    // ---- build fragments from LDS ----
    // 16-bit A layout: lanes 0-15 hold K=[h..h+7],[h+16..h+23] with h=0,
    // lanes 16-31 the same rows with h=8. (chunk1 = chunk0 + 16 K-elems)
    const int koff_u = (lane >> 4) * 4;  // K half-offset in uints (8 bf16)

    FragU afr[2];
#pragma unroll
    for (int i = 0; i < 2; ++i) {
      const unsigned* p = &xs[(wm + i * 16 + (lane & 15)) * ROW_U + koff_u];
      afr[i].q[0] = *(const uint4*)(p);
      afr[i].q[1] = *(const uint4*)(p + 8);
    }
    FragU bv[2], bt[2];
#pragma unroll
    for (int j = 0; j < 2; ++j) {
      const int row = (wn + j * 16 + (lane & 15)) * ROW_U + koff_u;
      const unsigned* pv = &wvs[row];
      bv[j].q[0] = *(const uint4*)(pv);
      bv[j].q[1] = *(const uint4*)(pv + 8);
      const unsigned* pt = &wts[row];
      bt[j].q[0] = *(const uint4*)(pt);
      bt[j].q[1] = *(const uint4*)(pt + 8);
    }

    // ---- 8 WMMAs: 2x2 subtiles x {W_v, W_t}, A fragments shared ----
#pragma unroll
    for (int i = 0; i < 2; ++i) {
#pragma unroll
      for (int j = 0; j < 2; ++j) {
        accV[i][j] = __builtin_amdgcn_wmma_f32_16x16x32_bf16(
            false, afr[i].v, false, bv[j].v, (short)0, accV[i][j], false, false);
        accT[i][j] = __builtin_amdgcn_wmma_f32_16x16x32_bf16(
            false, afr[i].v, false, bt[j].v, (short)0, accT[i][j], false, false);
      }
    }
    __syncthreads();
  }

  // ---- epilogue: per-token select via mask, store fp32 ----
  // C/D layout: lane -> N = lane&15 (+16 lanes repeat), VGPR r -> M = r + 8*(lane>=16)
#pragma unroll
  for (int i = 0; i < 2; ++i) {
    const int mbase = m0 + wm + i * 16 + ((lane >> 4) << 3);
#pragma unroll
    for (int j = 0; j < 2; ++j) {
      const int n = n0 + wn + j * 16 + (lane & 15);
      float* orow = out + (size_t)mbase * N_DIM + n;
#pragma unroll
      for (int r = 0; r < 8; ++r) {
        const bool mv = visual_mask[mbase + r] != 0;
        orow[(size_t)r * N_DIM] = mv ? accV[i][j][r] : accT[i][j][r];
      }
    }
  }
}

extern "C" void kernel_launch(void* const* d_in, const int* in_sizes, int n_in,
                              void* d_out, int out_size, void* d_ws, size_t ws_size,
                              hipStream_t stream) {
  (void)in_sizes; (void)n_in; (void)out_size; (void)d_ws; (void)ws_size;
  const float*   x    = (const float*)d_in[0];
  const uint8_t* mask = (const uint8_t*)d_in[1];   // jax bool -> 1 byte/elem
  const float*   Wv   = (const float*)d_in[2];
  const float*   Wt   = (const float*)d_in[3];
  float*         out  = (float*)d_out;

  dim3 grid(N_DIM / BN, M_TOT / BM);  // 16 x 256 blocks
  dim3 block(256);                    // 8 wave32s
  dual_gemm_masked_kernel<<<grid, block, 0, stream>>>(x, mask, Wv, Wt, out);
}
